// EquiConv_87608742903965
// MI455X (gfx1250) — matrix operand order repaired and checked by reading, not compiled, not run on hardware
//
#include <hip/hip_runtime.h>
#include <hip/hip_fp16.h>

typedef __attribute__((ext_vector_type(16))) _Float16 v16h;
typedef __attribute__((ext_vector_type(8)))  _Float16 v8h;
typedef __attribute__((ext_vector_type(2)))  _Float16 h2;
typedef __attribute__((ext_vector_type(8)))  float    v8f;

// constants folded into staged weights
#define C_SS0 0.02795084971874737f   // sqrt(1/1280)
#define C_PP0 0.01613743060919757f   // C_SS0/sqrt(3)
#define C_V1  0.03125f               // sqrt(3/1024)/sqrt(3)

// LDS layout (units: _Float16). Weights stored in WMMA B-fragment order:
// block(tile t, chunk c) = 512 halfs; lane L holds 16 contiguous halfs:
//   n = t*16 + (L&15), k = c*32 + 8*(L>>4) + (j&7) + 16*(j>>3) (j=0..15)
#define SG_BASE   0        // [1024 x 48] : 96 blocks (ss -> scalars|gates)
#define PP_BASE   49152    // [ 256 x 48] : 24 blocks (pp -> scalars|gates)
#define VEC_BASE  61440    // [1024 x 16] : 32 blocks (sp1 ; ps1)
#define FC1_BASE  77824    // [128 x 64]  : 16 blocks
#define FC2_BASE  86016    // [ 64 x 64]  :  8 blocks
#define FC3_BASE  90112    // [ 64 x 48]  :  6 blocks
#define SCR_BASE  93184    // 8 waves * (16 edges x 64 halfs) transpose scratch
#define LDS_HALFS 101376   // 202752 bytes (< 320KB WGP LDS)

__device__ __forceinline__ v8f wmma16(v16h a, v16h b, v8f c) {
  return __builtin_amdgcn_wmma_f32_16x16x32_f16(false, a, false, b, (short)0, c,
                                                false, false);
}
__device__ __forceinline__ h2 mk2(float a, float b) {
  h2 r; r.x = (_Float16)a; r.y = (_Float16)b; return r;
}
__device__ __forceinline__ h2 bc2(_Float16 s) { h2 r; r.x = s; r.y = s; return r; }
__device__ __forceinline__ void put2(v16h& a, int i, h2 t) { a[2*i] = t.x; a[2*i+1] = t.y; }
__device__ __forceinline__ v8f zero8() {
  v8f z;
#pragma unroll
  for (int i = 0; i < 8; ++i) z[i] = 0.f;
  return z;
}
__device__ __forceinline__ float siluf(float x) { return x / (1.f + __expf(-x)); }
__device__ __forceinline__ float sigmf(float x) { return 1.f / (1.f + __expf(-x)); }

__global__ __launch_bounds__(256) void equiconv_wmma(
    const float* __restrict__ in1,  const float* __restrict__ in2,
    const float* __restrict__ few,
    const float* __restrict__ wss0, const float* __restrict__ wpp0,
    const float* __restrict__ wss0g,const float* __restrict__ wpp0g,
    const float* __restrict__ wsp1, const float* __restrict__ wps1,
    const float* __restrict__ fcw1, const float* __restrict__ fcb1,
    const float* __restrict__ fcw2, const float* __restrict__ fcb2,
    const float* __restrict__ fcw3, const float* __restrict__ fcb3,
    float* __restrict__ out, int nTiles)
{
  extern __shared__ _Float16 sm[];
  const int tid = threadIdx.x;

  // ------------ stage weights f32 -> f16 in B-fragment order ------------
  for (int q = tid; q < 96*512; q += 256) {          // ss path, k = u*32+v
    int blk = q >> 9, r = q & 511, ln = r >> 4, j = r & 15;
    int t = blk >> 5, c = blk & 31;
    int k = c*32 + ((ln >> 4) << 3) + (j & 7) + ((j >> 3) << 4);
    int n = t*16 + (ln & 15);
    int u = k >> 5, v = k & 31;
    float val = (n < 32) ? C_SS0 * wss0[(u*32+v)*32 + n]
                         : C_SS0 * wss0g[(u*32+v)*16 + (n-32)];
    sm[SG_BASE + q] = (_Float16)val;
  }
  for (int q = tid; q < 24*512; q += 256) {          // pp path, k = u*16+v
    int blk = q >> 9, r = q & 511, ln = r >> 4, j = r & 15;
    int t = blk >> 3, c = blk & 7;
    int k = c*32 + ((ln >> 4) << 3) + (j & 7) + ((j >> 3) << 4);
    int n = t*16 + (ln & 15);
    int u = k >> 4, v = k & 15;
    float val = (n < 32) ? C_PP0 * wpp0[(u*16+v)*32 + n]
                         : C_PP0 * wpp0g[(u*16+v)*16 + (n-32)];
    sm[PP_BASE + q] = (_Float16)val;
  }
  for (int q = tid; q < 32*512; q += 256) {          // vec path: [sp1 ; ps1]
    int blk = q >> 9, r = q & 511, ln = r >> 4, j = r & 15;
    int k = blk*32 + ((ln >> 4) << 3) + (j & 7) + ((j >> 3) << 4);
    int n = ln & 15;
    float val = (k < 512) ? C_V1 * wsp1[k*16 + n] : C_V1 * wps1[(k-512)*16 + n];
    sm[VEC_BASE + q] = (_Float16)val;
  }
  for (int q = tid; q < 16*512; q += 256) {          // fc1 128x64
    int blk = q >> 9, r = q & 511, ln = r >> 4, j = r & 15;
    int t = blk >> 2, c = blk & 3;
    int k = c*32 + ((ln >> 4) << 3) + (j & 7) + ((j >> 3) << 4);
    int n = t*16 + (ln & 15);
    sm[FC1_BASE + q] = (_Float16)fcw1[k*64 + n];
  }
  for (int q = tid; q < 8*512; q += 256) {           // fc2 64x64
    int blk = q >> 9, r = q & 511, ln = r >> 4, j = r & 15;
    int t = blk >> 1, c = blk & 1;
    int k = c*32 + ((ln >> 4) << 3) + (j & 7) + ((j >> 3) << 4);
    int n = t*16 + (ln & 15);
    sm[FC2_BASE + q] = (_Float16)fcw2[k*64 + n];
  }
  for (int q = tid; q < 6*512; q += 256) {           // fc3 64x48
    int blk = q >> 9, r = q & 511, ln = r >> 4, j = r & 15;
    int t = blk >> 1, c = blk & 1;
    int k = c*32 + ((ln >> 4) << 3) + (j & 7) + ((j >> 3) << 4);
    int n = t*16 + (ln & 15);
    sm[FC3_BASE + q] = (_Float16)fcw3[k*48 + n];
  }
  __syncthreads();

  const int tile = blockIdx.x * 8 + (tid >> 5);      // one 16-edge tile per wave
  if (tile >= nTiles) return;

  const int lane = tid & 31;
  const int hb   = lane >> 4;    // K-half of the fragment this lane supplies
  const int nIdx = lane & 15;    // A-row (edge) index / D-column index
  const int e    = tile * 16 + nIdx;

  // ---------------- per-edge features -> f16 registers ------------------
  const float4* r1 = (const float4*)(in1 + (size_t)e * 80);
  const float4* r2 = (const float4*)(in2 + (size_t)e * 80);

  h2 s1[16];                                         // x1s as 16 packed pairs
#pragma unroll
  for (int k = 0; k < 8; ++k) {
    float4 f = r1[k];
    s1[2*k]   = mk2(f.x, f.y);
    s1[2*k+1] = mk2(f.z, f.w);
  }
  float xp1[48];
#pragma unroll
  for (int k = 0; k < 12; ++k) {
    float4 f = r1[8+k];
    xp1[4*k] = f.x; xp1[4*k+1] = f.y; xp1[4*k+2] = f.z; xp1[4*k+3] = f.w;
  }
  h2 p1[3][8];                                       // x1p[v,m] as v-pairs per m
#pragma unroll
  for (int m = 0; m < 3; ++m)
#pragma unroll
    for (int vp = 0; vp < 8; ++vp)
      p1[m][vp] = mk2(xp1[(2*vp)*3 + m], xp1[(2*vp+1)*3 + m]);

  float xs2[32];
#pragma unroll
  for (int k = 0; k < 8; ++k) {
    float4 f = r2[k];
    xs2[4*k] = f.x; xs2[4*k+1] = f.y; xs2[4*k+2] = f.z; xs2[4*k+3] = f.w;
  }
  float xp2[48];
#pragma unroll
  for (int k = 0; k < 12; ++k) {
    float4 f = r2[8+k];
    xp2[4*k] = f.x; xp2[4*k+1] = f.y; xp2[4*k+2] = f.z; xp2[4*k+3] = f.w;
  }
  // per-K-half views matching A-fragment layout (pair i: k=8h+2i / 16+8h+2(i-4))
  h2 sA[8];                                          // x2s run for this half
#pragma unroll
  for (int i = 0; i < 8; ++i) {
    int b0 = (i < 4) ? 2*i : 2*i + 8;
    sA[i] = hb ? mk2(xs2[b0+8], xs2[b0+9]) : mk2(xs2[b0], xs2[b0+1]);
  }
  h2 p2A[3][4];                                      // x2p v-pairs for this half
#pragma unroll
  for (int m = 0; m < 3; ++m)
#pragma unroll
    for (int j = 0; j < 4; ++j)
      p2A[m][j] = hb ? mk2(xp2[(8+2*j)*3 + m], xp2[(9+2*j)*3 + m])
                     : mk2(xp2[(2*j)*3 + m],   xp2[(2*j+1)*3 + m]);

  // ------------------------------- MLP ----------------------------------
  v8f hA[4];
#pragma unroll
  for (int t = 0; t < 4; ++t) hA[t] = zero8();
  const float* fr = few + (size_t)e * 128;
#pragma unroll
  for (int c = 0; c < 4; ++c) {                      // fc1: K=128
    int base = c*32 + hb*8;
    float4 q0 = *(const float4*)(fr + base);
    float4 q1 = *(const float4*)(fr + base + 4);
    float4 q2 = *(const float4*)(fr + base + 16);
    float4 q3 = *(const float4*)(fr + base + 20);
    v16h fa;
    put2(fa, 0, mk2(q0.x, q0.y)); put2(fa, 1, mk2(q0.z, q0.w));
    put2(fa, 2, mk2(q1.x, q1.y)); put2(fa, 3, mk2(q1.z, q1.w));
    put2(fa, 4, mk2(q2.x, q2.y)); put2(fa, 5, mk2(q2.z, q2.w));
    put2(fa, 6, mk2(q3.x, q3.y)); put2(fa, 7, mk2(q3.z, q3.w));
#pragma unroll
    for (int t = 0; t < 4; ++t)
      hA[t] = wmma16(fa, *(const v16h*)(sm + FC1_BASE + (t*4+c)*512 + lane*16), hA[t]);
  }
  _Float16* scr = sm + SCR_BASE + (tid >> 5) * 1024; // per-wave 16x64 scratch
#pragma unroll
  for (int t = 0; t < 4; ++t) {
    float b = fcb1[t*16 + nIdx];
#pragma unroll
    for (int r = 0; r < 8; ++r)
      scr[(r + 8*hb)*64 + t*16 + nIdx] = (_Float16)siluf(hA[t][r] + b);
  }
#pragma unroll
  for (int t = 0; t < 4; ++t) hA[t] = zero8();
#pragma unroll
  for (int c = 0; c < 2; ++c) {                      // fc2: K=64
    const _Float16* sp0 = scr + nIdx*64 + c*32 + hb*8;
    v8h lo = *(const v8h*)(sp0);
    v8h hi = *(const v8h*)(sp0 + 16);
    v16h fa;
#pragma unroll
    for (int i = 0; i < 8; ++i) { fa[i] = lo[i]; fa[8+i] = hi[i]; }
#pragma unroll
    for (int t = 0; t < 4; ++t)
      hA[t] = wmma16(fa, *(const v16h*)(sm + FC2_BASE + (t*2+c)*512 + lane*16), hA[t]);
  }
#pragma unroll
  for (int t = 0; t < 4; ++t) {
    float b = fcb2[t*16 + nIdx];
#pragma unroll
    for (int r = 0; r < 8; ++r)
      scr[(r + 8*hb)*64 + t*16 + nIdx] = (_Float16)siluf(hA[t][r] + b);
  }
  v8f wacc[3];
#pragma unroll
  for (int t = 0; t < 3; ++t) wacc[t] = zero8();
#pragma unroll
  for (int c = 0; c < 2; ++c) {                      // fc3: K=64
    const _Float16* sp0 = scr + nIdx*64 + c*32 + hb*8;
    v8h lo = *(const v8h*)(sp0);
    v8h hi = *(const v8h*)(sp0 + 16);
    v16h fa;
#pragma unroll
    for (int i = 0; i < 8; ++i) { fa[i] = lo[i]; fa[8+i] = hi[i]; }
#pragma unroll
    for (int t = 0; t < 3; ++t)
      wacc[t] = wmma16(fa, *(const v16h*)(sm + FC3_BASE + (t*2+c)*512 + lane*16), wacc[t]);
  }
#pragma unroll
  for (int t = 0; t < 3; ++t) {
    float b = fcb3[t*16 + nIdx];
#pragma unroll
    for (int r = 0; r < 8; ++r) wacc[t][r] += b;
  }

  // --------------- equivariant contractions (the big GEMMs) -------------
  v8f accSG[3];
#pragma unroll
  for (int t = 0; t < 3; ++t) accSG[t] = zero8();
  // ss: A[e, u*32+v] = x1s[u]*x2s[v], K=1024
#pragma unroll
  for (int c = 0; c < 32; ++c) {
    _Float16 x = (c & 1) ? s1[c >> 1].y : s1[c >> 1].x;
    h2 bc = bc2(x);
    v16h fa;
#pragma unroll
    for (int i = 0; i < 8; ++i) put2(fa, i, bc * sA[i]);
#pragma unroll
    for (int t = 0; t < 3; ++t)
      accSG[t] = wmma16(fa, *(const v16h*)(sm + SG_BASE + (t*32+c)*512 + lane*16), accSG[t]);
  }
  // pp: A[e, u*16+v] = sum_m x1p[u,m]*x2p[v,m], K=256
#pragma unroll
  for (int c = 0; c < 8; ++c) {
    h2 bl0 = bc2(p1[0][c].x), bl1 = bc2(p1[1][c].x), bl2 = bc2(p1[2][c].x);
    h2 bh0 = bc2(p1[0][c].y), bh1 = bc2(p1[1][c].y), bh2 = bc2(p1[2][c].y);
    v16h fa;
#pragma unroll
    for (int i = 0; i < 4; ++i) {
      put2(fa, i,     p2A[0][i]*bl0 + p2A[1][i]*bl1 + p2A[2][i]*bl2);
      put2(fa, i + 4, p2A[0][i]*bh0 + p2A[1][i]*bh1 + p2A[2][i]*bh2);
    }
#pragma unroll
    for (int t = 0; t < 3; ++t)
      accSG[t] = wmma16(fa, *(const v16h*)(sm + PP_BASE + (t*8+c)*512 + lane*16), accSG[t]);
  }
  // vec: per m, K=1024 = [sp1: x1s[u]*x2p[v,m] | ps1: x1p[u,m]*x2s[v]]
  v8f accV[3];
#pragma unroll
  for (int m = 0; m < 3; ++m) accV[m] = zero8();
#pragma unroll
  for (int m = 0; m < 3; ++m) {
#pragma unroll
    for (int c = 0; c < 16; ++c) {                   // sp1 half
      h2 blo = bc2(s1[c].x), bhi = bc2(s1[c].y);
      v16h fa;
#pragma unroll
      for (int i = 0; i < 4; ++i) {
        put2(fa, i,     blo * p2A[m][i]);
        put2(fa, i + 4, bhi * p2A[m][i]);
      }
      accV[m] = wmma16(fa, *(const v16h*)(sm + VEC_BASE + c*512 + lane*16), accV[m]);
    }
#pragma unroll
    for (int c = 0; c < 16; ++c) {                   // ps1 half
      _Float16 x = (c & 1) ? p1[m][c >> 1].y : p1[m][c >> 1].x;
      h2 bc = bc2(x);
      v16h fa;
#pragma unroll
      for (int i = 0; i < 8; ++i) put2(fa, i, bc * sA[i]);
      accV[m] = wmma16(fa, *(const v16h*)(sm + VEC_BASE + (16+c)*512 + lane*16), accV[m]);
    }
  }

  // ------------------------------ epilogue ------------------------------
  // D layout: (lane, reg r) = edge tile*16 + r + 8*hb, col = tileN*16 + nIdx
#pragma unroll
  for (int r = 0; r < 8; ++r) {
    size_t row = (size_t)(tile*16 + r + 8*hb) * 80;
    out[row + nIdx]      = siluf(accSG[0][r]) * wacc[0][r];
    out[row + 16 + nIdx] = siluf(accSG[1][r]) * wacc[1][r];
    float g = sigmf(accSG[2][r]) * wacc[2][r];
    out[row + 32 + nIdx*3 + 0] = g * accV[0][r];
    out[row + 32 + nIdx*3 + 1] = g * accV[1][r];
    out[row + 32 + nIdx*3 + 2] = g * accV[2][r];
  }
}

extern "C" void kernel_launch(void* const* d_in, const int* in_sizes, int n_in,
                              void* d_out, int out_size, void* d_ws, size_t ws_size,
                              hipStream_t stream) {
  (void)n_in; (void)out_size; (void)d_ws; (void)ws_size;
  const int E      = in_sizes[0] / 80;   // 100000
  const int nTiles = E / 16;             // 6250 (E divisible by 16)
  const int nBlk   = (nTiles + 7) / 8;   // 8 waves (tiles) per workgroup
  const size_t shm = (size_t)LDS_HALFS * sizeof(_Float16);  // ~198 KB
  hipLaunchKernelGGL(equiconv_wmma, dim3(nBlk), dim3(256), shm, stream,
      (const float*)d_in[0],  (const float*)d_in[1],  (const float*)d_in[2],
      (const float*)d_in[4],  (const float*)d_in[5],  (const float*)d_in[6],
      (const float*)d_in[7],  (const float*)d_in[8],  (const float*)d_in[9],
      (const float*)d_in[10], (const float*)d_in[11], (const float*)d_in[12],
      (const float*)d_in[13], (const float*)d_in[14], (const float*)d_in[15],
      (float*)d_out, nTiles);
}
